// MoEPred_51926154609359
// MI455X (gfx1250) — matrix operand order repaired and checked by reading, not compiled, not run on hardware
//
#include <hip/hip_runtime.h>

#define D_IN 512
#define HID 16
#define BATCH 262144
#define NT 34            // 32 expert tiles (16 hidden cols each) + 2 gating tiles
#define KSTEPS 16        // 512 / 32
#define PHASES 4
#define KS_PER_PHASE 4
#define ROWS_PER_BLOCK 128
#define THREADS 256
#define LDS_PHASE_DW (NT * KS_PER_PHASE * 256)   // 34816 dwords = 139264 bytes / phase

typedef __attribute__((ext_vector_type(16))) __bf16 v16bf;
typedef __attribute__((ext_vector_type(8)))  float  v8f;

#if __has_builtin(__builtin_amdgcn_global_load_async_to_lds_b128)
#define USE_ASYNC_LDS 1
#else
#define USE_ASYNC_LDS 0
#endif

// Exact pointee type from the builtin's signature: GCC vector_size(16) int
typedef int v4i_ __attribute__((vector_size(16)));
typedef __attribute__((address_space(1))) v4i_ gv4;   // global
typedef __attribute__((address_space(3))) v4i_ lv4;   // LDS

__device__ __forceinline__ lv4* as_lds(void* p) {
    // generic LDS pointer: low 32 bits are the LDS byte offset (aperture rule)
    return (lv4*)(unsigned int)(unsigned long long)p;
}
__device__ __forceinline__ gv4* as_glb(const void* p) {
    return (gv4*)(unsigned long long)p;
}

__device__ __forceinline__ unsigned int f32_to_bf16_bits(float f) {
    union { float f; unsigned int u; } c; c.f = f;
    unsigned int u = c.u;
    unsigned int r = u + 0x7FFFu + ((u >> 16) & 1u);   // round-to-nearest-even
    return r >> 16;
}

// ---------------------------------------------------------------------------
// Pack Wg[32,512] + W1[32,512,16] (f32) into bf16 WMMA B-operand layout.
// bp[((ks*NT + tile)*32 + lane)*8 + v] = packed {K=k0, K=k0+1} for col=lane&15
// k0 = ks*32 + (lane>>4)*16 + 2*v.  tile<32: expert `tile` from W1; 32/33: Wg.
// ---------------------------------------------------------------------------
__global__ void moe_pack_weights(const float* __restrict__ Wg,
                                 const float* __restrict__ W1,
                                 unsigned int* __restrict__ bp) {
    int idx  = blockIdx.x * blockDim.x + threadIdx.x;   // < 16*34*32*8 = 139264
    int v    = idx & 7;
    int lane = (idx >> 3) & 31;
    int tile = (idx >> 8) % NT;
    int ks   = idx / (NT * 256);
    int col  = lane & 15;
    int kh   = lane >> 4;
    int k0   = ks * 32 + kh * 16 + 2 * v;
    float f0, f1;
    if (tile < 32) {
        f0 = W1[(tile * D_IN + k0) * HID + col];
        f1 = W1[(tile * D_IN + k0 + 1) * HID + col];
    } else {
        int e = col + ((tile == 33) ? 16 : 0);
        f0 = Wg[e * D_IN + k0];
        f1 = Wg[e * D_IN + k0 + 1];
    }
    bp[idx] = f32_to_bf16_bits(f0) | (f32_to_bf16_bits(f1) << 16);
}

// ---------------------------------------------------------------------------
// Phase copy: 136KB of packed B -> LDS buffer (async-to-LDS when available)
// ---------------------------------------------------------------------------
__device__ __forceinline__ void issue_phase_copy(const unsigned int* src,
                                                 unsigned int* dst, int tid) {
#if USE_ASYNC_LDS
    gv4* s = as_glb(src);
    lv4* d = as_lds(dst);
    for (int i = tid; i < LDS_PHASE_DW / 4; i += THREADS)
        __builtin_amdgcn_global_load_async_to_lds_b128(s + i, d + i, 0, 0);
#else
    const uint4* s = (const uint4*)src;
    uint4* d = (uint4*)dst;
    for (int i = tid; i < LDS_PHASE_DW / 4; i += THREADS) d[i] = s[i];
#endif
}

__device__ __forceinline__ void wait_phase_copy() {
#if USE_ASYNC_LDS
#if __has_builtin(__builtin_amdgcn_s_wait_asynccnt)
    __builtin_amdgcn_s_wait_asynccnt(0);
#else
    asm volatile("s_wait_asynccnt 0x0" ::: "memory");
#endif
#endif
}

// ---------------------------------------------------------------------------
// Main fused kernel: 8 waves/block, 16 rows per wave, 34 WMMA column tiles,
// double-buffered async B staging.
// ---------------------------------------------------------------------------
__global__ __launch_bounds__(THREADS, 1)
void moe_main(const float* __restrict__ x,
              const unsigned int* __restrict__ bp,
              const float* __restrict__ b1,     // [32*16]
              const float* __restrict__ W2,     // [32*16]
              const float* __restrict__ b2,     // [32]
              float* __restrict__ out) {
    extern __shared__ unsigned int smem[];      // 2 x 139264 bytes
    const int tid  = threadIdx.x;
    const int lane = tid & 31;
    const int wave = tid >> 5;
    const int col  = lane & 15;
    const int kh   = lane >> 4;
    const int rowBase = blockIdx.x * ROWS_PER_BLOCK + wave * 16;

    v8f acc[NT];
    const v8f vzero = {0.f, 0.f, 0.f, 0.f, 0.f, 0.f, 0.f, 0.f};
#pragma unroll
    for (int t = 0; t < NT; ++t) acc[t] = vzero;

    const float* xrow = x + (size_t)(rowBase + col) * D_IN;   // A row = lane%16

    issue_phase_copy(bp, smem, tid);            // prologue: phase 0 -> buf 0

    for (int p = 0; p < PHASES; ++p) {
        wait_phase_copy();
        __syncthreads();                        // buf[p&1] ready for everyone
        if (p + 1 < PHASES)                     // overlap next copy with compute
            issue_phase_copy(bp + (size_t)(p + 1) * LDS_PHASE_DW,
                             smem + ((p + 1) & 1) * LDS_PHASE_DW, tid);

        const unsigned int* bufp = smem + (p & 1) * LDS_PHASE_DW;
        for (int ksl = 0; ksl < KS_PER_PHASE; ++ksl) {
            const int kbase = (p * KS_PER_PHASE + ksl) * 32;
            // A operand: lane holds K = kbase+kh*16+{0..15}
            const float4* xa = (const float4*)(xrow + kbase + kh * 16);
            float4 f0 = xa[0], f1 = xa[1], f2 = xa[2], f3 = xa[3];
            v16bf a;
            a[0]=(__bf16)f0.x;  a[1]=(__bf16)f0.y;  a[2]=(__bf16)f0.z;  a[3]=(__bf16)f0.w;
            a[4]=(__bf16)f1.x;  a[5]=(__bf16)f1.y;  a[6]=(__bf16)f1.z;  a[7]=(__bf16)f1.w;
            a[8]=(__bf16)f2.x;  a[9]=(__bf16)f2.y;  a[10]=(__bf16)f2.z; a[11]=(__bf16)f2.w;
            a[12]=(__bf16)f3.x; a[13]=(__bf16)f3.y; a[14]=(__bf16)f3.z; a[15]=(__bf16)f3.w;

            const unsigned int* bphase = bufp + ksl * NT * 256 + lane * 8;
#pragma unroll
            for (int t = 0; t < NT; ++t) {
                v16bf b = *(const v16bf*)(bphase + t * 256);
                acc[t] = __builtin_amdgcn_wmma_f32_16x16x32_bf16(
                    false, a, false, b, (short)0, acc[t], false, false);
            }
        }
        __syncthreads();                        // readers done before buf reuse
    }

    // ---------------- epilogue (per-wave private slab in buf 0) ------------
    // C layout: VGPR j, lane -> row r = j + 8*kh, col = lane&15
    float* post = (float*)smem + wave * 1024;   // 16 rows x 64 f32: [0..31]=out_e, [32..63]=logit
#pragma unroll
    for (int t = 0; t < 32; ++t) {
        float b1v = b1[t * HID + col];
        float w2v = W2[t * HID + col];
#pragma unroll
        for (int j = 0; j < 8; ++j) {
            float xv = acc[t][j] + b1v;
            float g  = 0.5f * xv * (1.f + tanhf(0.79788456f * (xv + 0.044715f * xv * xv * xv)));
            float s  = g * w2v;
#pragma unroll
            for (int msk = 1; msk < 16; msk <<= 1) s += __shfl_xor(s, msk, 32);
            if (col == (t & 15)) post[(j + 8 * kh) * 64 + t] = s;   // 2 lanes cover rows j, j+8
        }
    }
#pragma unroll
    for (int g = 0; g < 2; ++g) {              // gating logits, tiles 32/33
        int e = col + g * 16;
#pragma unroll
        for (int j = 0; j < 8; ++j)
            post[(j + 8 * kh) * 64 + 32 + e] = acc[32 + g][j];
    }
    if (lane < 16) {                           // lane = row within wave's 16 rows
        const float* pr = post + lane * 64;
        float mx = -1e30f;
#pragma unroll
        for (int e = 0; e < 32; ++e) mx = fmaxf(mx, pr[32 + e]);
        float num = 0.f, den = 0.f;
#pragma unroll
        for (int e = 0; e < 32; ++e) {
            float w = __expf(pr[32 + e] - mx);
            den += w;
            num += w * (pr[e] + b2[e]);
        }
        out[rowBase + lane] = num / den;
    }
}

extern "C" void kernel_launch(void* const* d_in, const int* in_sizes, int n_in,
                              void* d_out, int out_size, void* d_ws, size_t ws_size,
                              hipStream_t stream) {
    const float* x  = (const float*)d_in[0];
    const float* Wg = (const float*)d_in[1];
    const float* W1 = (const float*)d_in[2];
    const float* b1 = (const float*)d_in[3];
    const float* W2 = (const float*)d_in[4];
    const float* b2 = (const float*)d_in[5];
    float* out = (float*)d_out;
    unsigned int* bp = (unsigned int*)d_ws;    // needs 557056 bytes

    const int packElems = KSTEPS * NT * 32 * 8;   // 139264
    moe_pack_weights<<<packElems / 256, 256, 0, stream>>>(Wg, W1, bp);

    const size_t smemBytes = (size_t)2 * LDS_PHASE_DW * 4;  // 278528 B
    (void)hipFuncSetAttribute(reinterpret_cast<const void*>(moe_main),
                              hipFuncAttributeMaxDynamicSharedMemorySize, (int)smemBytes);
    moe_main<<<BATCH / ROWS_PER_BLOCK, THREADS, smemBytes, stream>>>(x, bp, b1, W2, b2, out);
}
